// GATEncoder_12867722018829
// MI455X (gfx1250) — matrix operand order repaired and checked by reading, not compiled, not run on hardware
//
#include <hip/hip_runtime.h>
#include <hip/hip_bf16.h>

#define NN  50000
#define EE  800000
#define HEADS 4
#define SLOPE 0.2f
#define ROWS 80          // rows per block: 50000 = 625 * 80

typedef __attribute__((ext_vector_type(16))) __bf16 bf16x16;
typedef __attribute__((ext_vector_type(8)))  __bf16 bf16x8;
typedef __attribute__((ext_vector_type(4)))  __bf16 bf16x4;
typedef __attribute__((ext_vector_type(8)))  float  f32x8;

__device__ __forceinline__ bf16x16 load_A_frag(const __bf16* ap) {
    // lane's A fragment: K = hi*8 + {0..7} and hi*8 + 16 + {0..7}  (two contiguous 16B runs)
    bf16x8 a0 = *(const bf16x8*)ap;
    bf16x8 a1 = *(const bf16x8*)(ap + 16);
    return __builtin_shufflevector(a0, a1, 0, 1, 2, 3, 4, 5, 6, 7,
                                   8, 9, 10, 11, 12, 13, 14, 15);
}

// ---------------- GEMM 1: hpre[N,128] = x[N,128] @ W1[128,128] (bf16 WMMA) ----------------
// 256 threads = 8 waves; wave w owns col-tile w; block owns an 80-row strip (5 row tiles).
__global__ __launch_bounds__(256) void gemm1_wmma(const float* __restrict__ x,
                                                  const float* __restrict__ W,
                                                  float* __restrict__ hout) {
    __shared__ __bf16 sx[ROWS * 128];   // [r][k]
    __shared__ __bf16 swT[128 * 128];   // [j][k]  (transposed weight panel)
    const int t = threadIdx.x;
    const size_t row0 = (size_t)blockIdx.x * ROWS;

    // stage x strip: float4 global loads -> packed bf16x4 LDS stores
    for (int i = t; i < ROWS * 32; i += 256) {
        float4 v = ((const float4*)x)[row0 * 32 + i];
        bf16x4 p;
        p[0] = (__bf16)v.x; p[1] = (__bf16)v.y; p[2] = (__bf16)v.z; p[3] = (__bf16)v.w;
        ((bf16x4*)sx)[i] = p;
    }
    // stage W transposed: thread i handles (k = i>>5, j = 4*(i&31))
    for (int i = t; i < 128 * 32; i += 256) {
        int k = i >> 5, j = (i & 31) * 4;
        float4 v = ((const float4*)W)[i];
        swT[(j + 0) * 128 + k] = (__bf16)v.x;
        swT[(j + 1) * 128 + k] = (__bf16)v.y;
        swT[(j + 2) * 128 + k] = (__bf16)v.z;
        swT[(j + 3) * 128 + k] = (__bf16)v.w;
    }
    __syncthreads();

    const int wave = t >> 5, lane = t & 31;
    const int col0 = wave * 16;
    const int mn   = lane & 15;
    const int hi   = lane >> 4;

    // hoist B fragments (reused across all 5 row tiles): K = hi*8 + {0..7} and hi*8 + 16 + {0..7} per chunk
    bf16x16 Bf[4];
#pragma unroll
    for (int c = 0; c < 4; ++c)
        Bf[c] = load_A_frag(&swT[(col0 + mn) * 128 + c * 32 + hi * 8]);

    for (int rt = 0; rt < ROWS / 16; ++rt) {
        f32x8 acc = {};
#pragma unroll
        for (int c = 0; c < 4; ++c) {
            bf16x16 A = load_A_frag(&sx[(rt * 16 + mn) * 128 + c * 32 + hi * 8]);
            acc = __builtin_amdgcn_wmma_f32_16x16x32_bf16(false, A, false, Bf[c],
                                                          (short)0, acc, false, false);
        }
#pragma unroll
        for (int v = 0; v < 8; ++v)
            hout[(row0 + rt * 16 + v + hi * 8) * 128 + col0 + mn] = acc[v];
    }
}

// ---------------- GEMM 2: h2[N,64] = h1[N,128] @ [W_mu | W_lv] (bf16 WMMA) ----------------
// 128 threads = 4 waves; wave w owns col-tile w of 64 cols; 80-row strip per block.
__global__ __launch_bounds__(128) void gemm2_wmma(const float* __restrict__ h1,
                                                  const float* __restrict__ Wmu,
                                                  const float* __restrict__ Wlv,
                                                  float* __restrict__ h2) {
    __shared__ __bf16 sx[ROWS * 128];   // [r][k]
    __shared__ __bf16 swT[64 * 128];    // [j][k]; j<32 mu, j>=32 lv
    const int t = threadIdx.x;
    const size_t row0 = (size_t)blockIdx.x * ROWS;

    for (int i = t; i < ROWS * 32; i += 128) {
        float4 v = ((const float4*)h1)[row0 * 32 + i];
        bf16x4 p;
        p[0] = (__bf16)v.x; p[1] = (__bf16)v.y; p[2] = (__bf16)v.z; p[3] = (__bf16)v.w;
        ((bf16x4*)sx)[i] = p;
    }
    // stage both weights transposed: thread i -> (k = i>>3, j = 4*(i&7))
    for (int i = t; i < 128 * 8; i += 128) {
        int k = i >> 3, j = (i & 7) * 4;
        float4 vm = ((const float4*)Wmu)[i];
        float4 vl = ((const float4*)Wlv)[i];
        swT[(j + 0) * 128 + k] = (__bf16)vm.x;
        swT[(j + 1) * 128 + k] = (__bf16)vm.y;
        swT[(j + 2) * 128 + k] = (__bf16)vm.z;
        swT[(j + 3) * 128 + k] = (__bf16)vm.w;
        swT[(32 + j + 0) * 128 + k] = (__bf16)vl.x;
        swT[(32 + j + 1) * 128 + k] = (__bf16)vl.y;
        swT[(32 + j + 2) * 128 + k] = (__bf16)vl.z;
        swT[(32 + j + 3) * 128 + k] = (__bf16)vl.w;
    }
    __syncthreads();

    const int wave = t >> 5, lane = t & 31;
    const int col0 = wave * 16;
    const int mn   = lane & 15;
    const int hi   = lane >> 4;

    bf16x16 Bf[4];
#pragma unroll
    for (int c = 0; c < 4; ++c)
        Bf[c] = load_A_frag(&swT[(col0 + mn) * 128 + c * 32 + hi * 8]);

    for (int rt = 0; rt < ROWS / 16; ++rt) {
        f32x8 acc = {};
#pragma unroll
        for (int c = 0; c < 4; ++c) {
            bf16x16 A = load_A_frag(&sx[(rt * 16 + mn) * 128 + c * 32 + hi * 8]);
            acc = __builtin_amdgcn_wmma_f32_16x16x32_bf16(false, A, false, Bf[c],
                                                          (short)0, acc, false, false);
        }
#pragma unroll
        for (int v = 0; v < 8; ++v)
            h2[(row0 + rt * 16 + v + hi * 8) * 64 + col0 + mn] = acc[v];
    }
}

// ------------- monotonic float<->uint key for order-independent atomic max -------------
__device__ __forceinline__ unsigned f2key(float f) {
    unsigned b = __float_as_uint(f);
    return (b & 0x80000000u) ? ~b : (b | 0x80000000u);
}
__device__ __forceinline__ float key2f(unsigned k) {
    unsigned b = (k & 0x80000000u) ? (k & 0x7FFFFFFFu) : ~k;
    return __uint_as_float(b);
}

// ---------------- Layer-1 attention dots + init accumulators ----------------
__global__ void att1_init(const float* __restrict__ hpre,
                          const float* __restrict__ att_s, const float* __restrict__ att_d,
                          float* __restrict__ a_src, float* __restrict__ a_dst,
                          unsigned* __restrict__ emaxk, float* __restrict__ den,
                          float* __restrict__ out1) {
    int i = blockIdx.x * blockDim.x + threadIdx.x;
    if (i < NN * 128) out1[i] = 0.0f;
    if (i < NN * HEADS) {
        int n = i >> 2, h = i & 3;
        const float* hp = hpre + (size_t)n * 128 + h * 32;
        float s = 0.f, d = 0.f;
#pragma unroll 8
        for (int k = 0; k < 32; ++k) {
            float v = hp[k];
            s += v * att_s[h * 32 + k];
            d += v * att_d[h * 32 + k];
        }
        a_src[i] = s;
        a_dst[i] = d;
        emaxk[i] = 0u;   // key 0 == smallest float
        den[i]   = 0.f;
    }
}

// ---------------- Layer-1 edge passes ----------------
__global__ void edge_max1(const int* __restrict__ src, const int* __restrict__ dst,
                          const float* __restrict__ a_src, const float* __restrict__ a_dst,
                          unsigned* __restrict__ emaxk) {
    int i = blockIdx.x * blockDim.x + threadIdx.x;
    if (i >= EE * HEADS) return;
    int e = i >> 2, h = i & 3;
    int s = src[e], d = dst[e];
    float v = a_src[s * 4 + h] + a_dst[d * 4 + h];
    v = (v > 0.f) ? v : SLOPE * v;
    atomicMax(&emaxk[d * 4 + h], f2key(v));
}

__global__ void edge_exp1(const int* __restrict__ src, const int* __restrict__ dst,
                          const float* __restrict__ a_src, const float* __restrict__ a_dst,
                          const unsigned* __restrict__ emaxk, float* __restrict__ den,
                          float* __restrict__ eexp) {
    int i = blockIdx.x * blockDim.x + threadIdx.x;
    if (i >= EE * HEADS) return;
    int e = i >> 2, h = i & 3;
    int s = src[e], d = dst[e];
    float v = a_src[s * 4 + h] + a_dst[d * 4 + h];
    v = (v > 0.f) ? v : SLOPE * v;
    float ex = __expf(v - key2f(emaxk[d * 4 + h]));
    eexp[i] = ex;
    atomicAdd(&den[d * 4 + h], ex);
}

// thread per (edge, float4-chunk): 32 chunks cover 128 dims; head = chunk>>3
__global__ void edge_aggr1(const int* __restrict__ src, const int* __restrict__ dst,
                           const float* __restrict__ hpre, const float* __restrict__ eexp,
                           const float* __restrict__ den, float* __restrict__ out1) {
    long long i = (long long)blockIdx.x * blockDim.x + threadIdx.x;
    if (i >= (long long)EE * 32) return;
    int e = (int)(i >> 5), q = (int)(i & 31);
    int h = q >> 3;
    int s = src[e], d = dst[e];
    float alpha = eexp[e * 4 + h] / (den[d * 4 + h] + 1e-16f);
    const float4 hv = *(const float4*)(hpre + (size_t)s * 128 + q * 4);
    float* o = out1 + (size_t)d * 128 + q * 4;
    atomicAdd(o + 0, hv.x * alpha);
    atomicAdd(o + 1, hv.y * alpha);
    atomicAdd(o + 2, hv.z * alpha);
    atomicAdd(o + 3, hv.w * alpha);
}

// ---------------- ELU + bias (in place: out1 -> h1) ----------------
__global__ void elu_bias1(float* __restrict__ out1, const float* __restrict__ b1) {
    int i = blockIdx.x * blockDim.x + threadIdx.x;
    if (i >= NN * 128) return;
    float v = out1[i] + b1[i & 127];
    out1[i] = (v > 0.f) ? v : (__expf(v) - 1.0f);
}

// ---------------- Layer-2 attention dots (mu & lv) + init + zero d_out ----------------
__global__ void att2_init(const float* __restrict__ h2,
                          const float* __restrict__ ams, const float* __restrict__ amd_,
                          const float* __restrict__ als, const float* __restrict__ ald,
                          float* __restrict__ a2, unsigned* __restrict__ emaxk,
                          float* __restrict__ den, float* __restrict__ dout) {
    int i = blockIdx.x * blockDim.x + threadIdx.x;
    if (i < NN * 64) dout[i] = 0.0f;    // d_out = [mu N*32 | lv N*32]
    if (i < NN) {
        const float* p = h2 + (size_t)i * 64;
        float smu = 0.f, dmu = 0.f, slv = 0.f, dlv = 0.f;
#pragma unroll 8
        for (int k = 0; k < 32; ++k) {
            float vm = p[k], vl = p[32 + k];
            smu += vm * ams[k];
            dmu += vm * amd_[k];
            slv += vl * als[k];
            dlv += vl * ald[k];
        }
        a2[i * 4 + 0] = smu;
        a2[i * 4 + 1] = dmu;
        a2[i * 4 + 2] = slv;
        a2[i * 4 + 3] = dlv;
        emaxk[i * 2 + 0] = 0u;
        emaxk[i * 2 + 1] = 0u;
        den[i * 2 + 0] = 0.f;
        den[i * 2 + 1] = 0.f;
    }
}

__global__ void edge_max2(const int* __restrict__ src, const int* __restrict__ dst,
                          const float* __restrict__ a2, unsigned* __restrict__ emaxk) {
    int i = blockIdx.x * blockDim.x + threadIdx.x;
    if (i >= EE * 2) return;
    int e = i >> 1, c = i & 1;    // 0=mu, 1=lv
    int s = src[e], d = dst[e];
    float v = a2[s * 4 + 2 * c] + a2[d * 4 + 2 * c + 1];
    v = (v > 0.f) ? v : SLOPE * v;
    atomicMax(&emaxk[d * 2 + c], f2key(v));
}

__global__ void edge_exp2(const int* __restrict__ src, const int* __restrict__ dst,
                          const float* __restrict__ a2, const unsigned* __restrict__ emaxk,
                          float* __restrict__ den, float* __restrict__ eexp) {
    int i = blockIdx.x * blockDim.x + threadIdx.x;
    if (i >= EE * 2) return;
    int e = i >> 1, c = i & 1;
    int s = src[e], d = dst[e];
    float v = a2[s * 4 + 2 * c] + a2[d * 4 + 2 * c + 1];
    v = (v > 0.f) ? v : SLOPE * v;
    float ex = __expf(v - key2f(emaxk[d * 2 + c]));
    eexp[i] = ex;
    atomicAdd(&den[d * 2 + c], ex);
}

// thread per (edge, q), q in 0..15: q<8 -> mu float4 chunks, q>=8 -> lv
__global__ void edge_aggr2(const int* __restrict__ src, const int* __restrict__ dst,
                           const float* __restrict__ h2, const float* __restrict__ eexp,
                           const float* __restrict__ den, float* __restrict__ dout) {
    long long i = (long long)blockIdx.x * blockDim.x + threadIdx.x;
    if (i >= (long long)EE * 16) return;
    int e = (int)(i >> 4), q = (int)(i & 15);
    int c = q >> 3, d4 = q & 7;
    int s = src[e], d = dst[e];
    float alpha = eexp[e * 2 + c] / (den[d * 2 + c] + 1e-16f);
    const float4 hv = *(const float4*)(h2 + (size_t)s * 64 + c * 32 + d4 * 4);
    float* o = dout + (size_t)c * NN * 32 + (size_t)d * 32 + d4 * 4;
    atomicAdd(o + 0, hv.x * alpha);
    atomicAdd(o + 1, hv.y * alpha);
    atomicAdd(o + 2, hv.z * alpha);
    atomicAdd(o + 3, hv.w * alpha);
}

__global__ void bias2(float* __restrict__ dout, const float* __restrict__ bmu,
                      const float* __restrict__ blv) {
    int i = blockIdx.x * blockDim.x + threadIdx.x;
    if (i >= NN * 64) return;
    int d = i & 31;
    dout[i] += (i < NN * 32) ? bmu[d] : blv[d];
}

extern "C" void kernel_launch(void* const* d_in, const int* in_sizes, int n_in,
                              void* d_out, int out_size, void* d_ws, size_t ws_size,
                              hipStream_t stream) {
    const float* x    = (const float*)d_in[0];
    const int*   src  = (const int*)d_in[1];
    const int*   dst  = (const int*)d_in[2];
    const float* W1   = (const float*)d_in[3];
    const float* a1s  = (const float*)d_in[4];
    const float* a1d  = (const float*)d_in[5];
    const float* b1   = (const float*)d_in[6];
    const float* Wmu  = (const float*)d_in[7];
    const float* amus = (const float*)d_in[8];
    const float* amud = (const float*)d_in[9];
    const float* bmu  = (const float*)d_in[10];
    const float* Wlv  = (const float*)d_in[11];
    const float* alvs = (const float*)d_in[12];
    const float* alvd = (const float*)d_in[13];
    const float* blv  = (const float*)d_in[14];
    float*       dout = (float*)d_out;

    // workspace layout (floats); layer-2 buffers alias dead layer-1 buffers
    float* ws     = (float*)d_ws;
    float* hpre   = ws;                          // N*128
    float* out1   = hpre + (size_t)NN * 128;     // N*128 (becomes h1 after elu_bias1)
    float* small  = out1 + (size_t)NN * 128;     // N*16 region
    float* a_src1 = small;                       // N*4
    float* a_dst1 = small + (size_t)NN * 4;      // N*4
    unsigned* emax1k = (unsigned*)(small + (size_t)NN * 8);  // N*4
    float* den1   = small + (size_t)NN * 12;     // N*4
    float* eexp1  = small + (size_t)NN * 16;     // E*4
    // layer 2 aliases
    float* h2     = hpre;                        // N*64 <= N*128
    float* a2     = small;                       // N*4
    unsigned* emax2k = (unsigned*)(small + (size_t)NN * 4);  // N*2
    float* den2   = small + (size_t)NN * 6;      // N*2
    float* eexp2  = eexp1;                       // E*2 <= E*4

    const int B = 256;

    // ---- layer 1 ----
    gemm1_wmma<<<NN / ROWS, 256, 0, stream>>>(x, W1, hpre);
    att1_init<<<(NN * 128 + B - 1) / B, B, 0, stream>>>(hpre, a1s, a1d, a_src1, a_dst1,
                                                        emax1k, den1, out1);
    edge_max1<<<(EE * 4 + B - 1) / B, B, 0, stream>>>(src, dst, a_src1, a_dst1, emax1k);
    edge_exp1<<<(EE * 4 + B - 1) / B, B, 0, stream>>>(src, dst, a_src1, a_dst1, emax1k,
                                                      den1, eexp1);
    edge_aggr1<<<(int)(((long long)EE * 32 + B - 1) / B), B, 0, stream>>>(src, dst, hpre,
                                                                          eexp1, den1, out1);
    elu_bias1<<<(NN * 128 + B - 1) / B, B, 0, stream>>>(out1, b1);

    // ---- layer 2 (mu & logvar fused) ----
    gemm2_wmma<<<NN / ROWS, 128, 0, stream>>>(out1, Wmu, Wlv, h2);
    att2_init<<<(NN * 64 + B - 1) / B, B, 0, stream>>>(h2, amus, amud, alvs, alvd, a2,
                                                       emax2k, den2, dout);
    edge_max2<<<(EE * 2 + B - 1) / B, B, 0, stream>>>(src, dst, a2, emax2k);
    edge_exp2<<<(EE * 2 + B - 1) / B, B, 0, stream>>>(src, dst, a2, emax2k, den2, eexp2);
    edge_aggr2<<<(int)(((long long)EE * 16 + B - 1) / B), B, 0, stream>>>(src, dst, h2,
                                                                          eexp2, den2, dout);
    bias2<<<(NN * 64 + B - 1) / B, B, 0, stream>>>(dout, bmu, blv);
}
